// Transformer_23244363006481
// MI455X (gfx1250) — matrix operand order repaired and checked by reading, not compiled, not run on hardware
//
#include <hip/hip_runtime.h>
#include <hip/hip_bf16.h>
#include <math.h>

typedef __attribute__((ext_vector_type(16))) _Float16     v16h;
typedef __attribute__((ext_vector_type(8)))  _Float16     v8h;
typedef __attribute__((ext_vector_type(8)))  float        v8f;
typedef __attribute__((ext_vector_type(4)))  unsigned int u32x4;
typedef __attribute__((ext_vector_type(8)))  int          i32x8;
typedef __attribute__((ext_vector_type(4)))  int          i32x4;

#define ROWS   2048   // B*T
#define D      64
#define TSEQ   256
#define NHEAD  8
#define HS     8
#define FFN    1024
#define NVOCAB 50257

// ---------------------------------------------------------------------------
// Tensor Data Mover: 2D tile load Global -> LDS (f16 elements).
// D# per CDNA5 ISA §8.3/8.4. OOB rows (row >= dim1) read as zero ->
// the ragged vocab edge is zero-filled for free.
// ---------------------------------------------------------------------------
__device__ __forceinline__ void tdm_load_2d_f16(const void* gptr,
                                                unsigned lds_byte_off,
                                                unsigned dim0,      // tensor width (elems)
                                                unsigned dim1,      // tensor rows remaining
                                                unsigned tile0,     // tile width (elems)
                                                unsigned tile1,     // tile rows
                                                unsigned stride0) { // row stride (elems)
    unsigned long long ga = (unsigned long long)(uintptr_t)gptr;
    u32x4 g0 = {0u, 0u, 0u, 0u};
    g0[0] = 1u;                                   // count=1, user mode
    g0[1] = lds_byte_off;                         // lds_addr
    g0[2] = (unsigned)ga;                         // global_addr[31:0]
    g0[3] = (unsigned)((ga >> 32) & 0x01FFFFFFu)  // global_addr[56:32]
            | (2u << 30);                         // type = 2 ("image")
    i32x8 g1 = {0, 0, 0, 0, 0, 0, 0, 0};
    g1[0] = (int)(1u << 16);                      // data_size=1 (2 bytes)
    g1[1] = (int)((dim0 & 0xFFFFu) << 16);        // tensor_dim0[15:0]
    g1[2] = (int)((dim0 >> 16) | ((dim1 & 0xFFFFu) << 16)); // dim0[31:16]|dim1[15:0]
    g1[3] = (int)((dim1 >> 16) | ((tile0 & 0xFFFFu) << 16));// dim1[31:16]|tile_dim0
    g1[4] = (int)(tile1 & 0xFFFFu);               // tile_dim1 (tile_dim2=0)
    g1[5] = (int)stride0;                         // tensor_dim0_stride[31:0]
    i32x4 gz = {0, 0, 0, 0};
#if __clang_major__ >= 23
    i32x8 gz8 = {0, 0, 0, 0, 0, 0, 0, 0};
    __builtin_amdgcn_tensor_load_to_lds(g0, g1, gz, gz, gz8, 0);
#else
    __builtin_amdgcn_tensor_load_to_lds(g0, g1, gz, gz, 0);
#endif
}

// ---------------------------------------------------------------------------
// Embedding gather + sinusoidal positional encoding (fp32 + f16 shadow)
// ---------------------------------------------------------------------------
__global__ void embed_kernel(const int* __restrict__ tokens,
                             const float* __restrict__ emb,
                             float* __restrict__ x,
                             _Float16* __restrict__ xh) {
    int idx = blockIdx.x * blockDim.x + threadIdx.x;   // over ROWS*D
    if (idx >= ROWS * D) return;
    int row = idx >> 6;
    int c   = idx & 63;
    int t   = row & (TSEQ - 1);
    int tok = tokens[row];
    float i2  = (float)(c & ~1);
    float ang = (float)t * __expf(-9.210340371976184f * i2 * (1.0f / 64.0f));
    float pe  = (c & 1) ? __cosf(ang) : __sinf(ang);
    float val = emb[tok * D + c] + pe;
    x[idx]  = val;
    xh[idx] = (_Float16)val;
}

// ---------------------------------------------------------------------------
// Weight pre-pass: W[K,N] fp32 row-major  ->  WT[N,K] f16 (contiguous K runs)
// ---------------------------------------------------------------------------
__global__ void transpose_f16_kernel(const float* __restrict__ W,
                                     _Float16* __restrict__ WT,
                                     int K, int N) {
    int idx = blockIdx.x * blockDim.x + threadIdx.x;
    if (idx >= N * K) return;
    int n = idx / K;
    int k = idx - n * K;
    WT[idx] = (_Float16)W[(size_t)k * N + n];
}

// ---------------------------------------------------------------------------
// GEMM: C[2048,N] = A[2048,K]*B[K,N] + bias (opt ReLU).
// A is f16 row-major; B is pre-transposed BT[N,K] f16. N,K compile-time so
// store row strides fold into immediate offsets. Block = 8 waves = 8 M-tiles
// sharing one 16-wide N tile whose 16xK f16 slab is staged into LDS by the
// Tensor Data Mover (zero-filled past row N). Per-wave v_wmma_f32_16x16x32_f16:
//   A (16x32): lane<16 -> M=lane,  K={0..7}u{16..23}; lane>=16 -> M=lane-16, K+8
//   B (32x16): lane<16 -> N=lane,  K={0..7}u{16..23}; lane>=16 -> N=lane-16, K+8
//   C tile:    vgpr r:  lane<16 -> (M=r, N=lane); lane>=16 -> (M=r+8, N=l-16)
// ---------------------------------------------------------------------------
template <int K, int N, bool RELU, bool OUTH>
__global__ void gemm_wmma_tdm_kernel(const _Float16* __restrict__ A,
                                     const _Float16* __restrict__ BT,
                                     const float* __restrict__ bias,
                                     float* __restrict__ Cf,
                                     _Float16* __restrict__ Ch) {
    __shared__ __align__(16) _Float16 bt[16 * K];   // 2KB (K=64) / 32KB (K=1024)

    const int nt     = blockIdx.x >> 4;     // N-tile index
    const int mchunk = blockIdx.x & 15;     // chunk of 8 M-tiles (M=2048)
    const int n0     = nt << 4;

    if (threadIdx.x < 32) {                 // one wave issues the TDM
        tdm_load_2d_f16(BT + (size_t)n0 * K, 0u,
                        (unsigned)K, (unsigned)(N - n0),
                        (unsigned)K, 16u, (unsigned)K);
        __builtin_amdgcn_s_wait_tensorcnt(0);
    }
    __syncthreads();

    const int lane   = threadIdx.x & 31;
    const int wv     = threadIdx.x >> 5;
    const int m0     = (mchunk * 8 + wv) << 4;
    const int lane15 = lane & 15;
    const int hi     = lane >> 4;
    const int ka     = hi ? 8 : 0;
    const int kb     = hi ? 8 : 0;
    const int mrow   = m0 + lane15;

    v8f acc = {};
#pragma unroll 4
    for (int k0 = 0; k0 < K; k0 += 32) {
        // A fragment: two 16B f16 loads (no converts)
        const v8h* __restrict__ Ap =
            (const v8h*)(A + (size_t)mrow * K + k0 + ka);
        v8h a0 = Ap[0], a1 = Ap[2];            // +16 halves
        // B fragment: two 16B LDS loads
        const v8h* __restrict__ Bl =
            (const v8h*)(bt + (size_t)lane15 * K + k0 + kb);
        v8h b0 = Bl[0], b1 = Bl[2];
        v16h a, b;
#pragma unroll
        for (int j = 0; j < 8; ++j) {
            a[j] = a0[j]; a[j + 8] = a1[j];
            b[j] = b0[j]; b[j + 8] = b1[j];
        }
        acc = __builtin_amdgcn_wmma_f32_16x16x32_f16(
            false, a, false, b, (short)0, acc, false, false);
    }

    const int ncol = n0 + lane15;
    if constexpr (N % 16 != 0) {
        if (ncol >= N) return;              // ragged vocab edge (stores only)
    }
    const float bb   = bias[ncol];
    const int   crow = m0 + (hi ? 8 : 0);
#pragma unroll
    for (int r = 0; r < 8; ++r) {
        float val = acc[r] + bb;
        if (RELU) val = fmaxf(val, 0.0f);
        if constexpr (OUTH)
            Ch[(size_t)(crow + r) * N + ncol] = (_Float16)val;
        else
            Cf[(size_t)(crow + r) * N + ncol] = val;
    }
}

// ---------------------------------------------------------------------------
// Causal attention: one (batch,head) per block, one query row per thread.
// head_size=8, T=256; K/V staged in LDS; online softmax. Output f16 (it is
// only ever consumed as a GEMM A-operand).
// ---------------------------------------------------------------------------
__global__ void attention_kernel(const float* __restrict__ q,
                                 const float* __restrict__ k,
                                 const float* __restrict__ v,
                                 _Float16* __restrict__ out) {
    const int bh = blockIdx.x;
    const int b  = bh >> 3;
    const int h  = bh & 7;
    const int t  = threadIdx.x;

    __shared__ float ks[TSEQ][HS];
    __shared__ float vs[TSEQ][HS];

    const size_t base = (size_t)b * TSEQ * D + (size_t)h * HS;
#pragma unroll
    for (int i = 0; i < HS; ++i) {
        ks[t][i] = k[base + (size_t)t * D + i];
        vs[t][i] = v[base + (size_t)t * D + i];
    }
    __syncthreads();

    float qr[HS];
#pragma unroll
    for (int i = 0; i < HS; ++i) qr[i] = q[base + (size_t)t * D + i];

    const float scale = 0.3535533905932738f;   // 1/sqrt(8)
    float mx = -3.4e38f, l = 0.0f;
    float acc[HS] = {0, 0, 0, 0, 0, 0, 0, 0};
    for (int s = 0; s <= t; ++s) {
        float d = 0.0f;
#pragma unroll
        for (int i = 0; i < HS; ++i) d += qr[i] * ks[s][i];
        d *= scale;
        float mn   = fmaxf(mx, d);
        float corr = __expf(mx - mn);
        float p    = __expf(d - mn);
        l = l * corr + p;
#pragma unroll
        for (int i = 0; i < HS; ++i) acc[i] = acc[i] * corr + p * vs[s][i];
        mx = mn;
    }
    const float inv = 1.0f / l;
#pragma unroll
    for (int i = 0; i < HS; ++i)
        out[base + (size_t)t * D + i] = (_Float16)(acc[i] * inv);
}

// ---------------------------------------------------------------------------
// out = LayerNorm(x [+ h]) * g + b ; one wave per 64-wide row, wave32 shuffle
// reductions. Optional f16 shadow copy for downstream GEMM A-operands.
// ---------------------------------------------------------------------------
__global__ void add_ln_kernel(const float* __restrict__ x,
                              const float* __restrict__ h,
                              const float* __restrict__ g,
                              const float* __restrict__ bta,
                              float* __restrict__ out,
                              _Float16* __restrict__ outh) {
    const int row  = blockIdx.x * 8 + (threadIdx.x >> 5);
    const int lane = threadIdx.x & 31;
    if (row >= ROWS) return;
    const size_t off = (size_t)row * D;

    float v0 = x[off + lane];
    float v1 = x[off + lane + 32];
    if (h) { v0 += h[off + lane]; v1 += h[off + lane + 32]; }

    float s = v0 + v1;
#pragma unroll
    for (int o = 16; o > 0; o >>= 1) s += __shfl_xor(s, o, 32);
    const float mu = s * (1.0f / 64.0f);

    const float d0 = v0 - mu, d1 = v1 - mu;
    float qv = d0 * d0 + d1 * d1;
#pragma unroll
    for (int o = 16; o > 0; o >>= 1) qv += __shfl_xor(qv, o, 32);
    const float rstd = rsqrtf(qv * (1.0f / 64.0f) + 1e-5f);

    const float r0 = d0 * rstd * g[lane]      + bta[lane];
    const float r1 = d1 * rstd * g[lane + 32] + bta[lane + 32];
    out[off + lane]      = r0;
    out[off + lane + 32] = r1;
    if (outh) {
        outh[off + lane]      = (_Float16)r0;
        outh[off + lane + 32] = (_Float16)r1;
    }
}

// ---------------------------------------------------------------------------
// Host-side orchestration
// ---------------------------------------------------------------------------
template <int K, int N, bool RELU, bool OUTH>
static inline void run_gemm(const _Float16* A, const _Float16* BT,
                            const float* bias, float* Cf, _Float16* Ch,
                            hipStream_t stream) {
    constexpr int blocks = ((N + 15) / 16) * 16;   // n-tiles x 16 m-chunks
    gemm_wmma_tdm_kernel<K, N, RELU, OUTH>
        <<<blocks, 256, 0, stream>>>(A, BT, bias, Cf, Ch);
}

static inline void launch_transpose(const float* W, _Float16* WT,
                                    int K, int N, hipStream_t stream) {
    const int total = K * N;
    transpose_f16_kernel<<<(total + 255) / 256, 256, 0, stream>>>(W, WT, K, N);
}

extern "C" void kernel_launch(void* const* d_in, const int* in_sizes, int n_in,
                              void* d_out, int out_size, void* d_ws, size_t ws_size,
                              hipStream_t stream) {
    (void)in_sizes; (void)n_in; (void)out_size; (void)ws_size;

    const int*   tokens = (const int*)d_in[0];
    const float* emb    = (const float*)d_in[1];
    auto P = [&](int layer, int j) -> const float* {
        return (const float*)d_in[2 + layer * 16 + j];
    };
    const float* lnf_g = (const float*)d_in[98];
    const float* lnf_b = (const float*)d_in[99];
    const float* Wout  = (const float*)d_in[100];
    const float* bout  = (const float*)d_in[101];
    float*       logits = (float*)d_out;

    // ---- workspace layout -------------------------------------------------
    float* ws  = (float*)d_ws;
    float* x   = ws;                  // [2048,64] fp32 residual stream
    float* xq  = ws + 1 * 131072;
    float* xk  = ws + 2 * 131072;
    float* xv  = ws + 3 * 131072;
    float* tmp = ws + 4 * 131072;
    // f16 region (byte offset 2,621,440 -> 16B aligned)
    _Float16* hb  = (_Float16*)(ws + 5 * 131072);
    _Float16* xh  = hb;                    // f16 shadow of x     [2048,64]
    _Float16* aoh = hb + 131072;           // attention out (f16) [2048,64]
    _Float16* xfh = hb + 262144;           // final-LN out (f16)  [2048,64]
    _Float16* h1h = hb + 393216;           // FFN hidden (f16)    [2048,1024]
    _Float16* wt  = hb + 393216 + ROWS * FFN;  // transposed weights
    _Float16* WTl[6][6];
    size_t hoff = 0;
    for (int L = 0; L < 6; ++L) {
        WTl[L][0] = wt + hoff; hoff += D * D;        // WqT  [64,64]
        WTl[L][1] = wt + hoff; hoff += D * D;        // WkT
        WTl[L][2] = wt + hoff; hoff += D * D;        // WvT
        WTl[L][3] = wt + hoff; hoff += D * D;        // WoT
        WTl[L][4] = wt + hoff; hoff += FFN * D;      // W1T  [1024,64]
        WTl[L][5] = wt + hoff; hoff += D * FFN;      // W2T  [64,1024]
    }
    _Float16* WoutT = wt + hoff;                     // [50257,64]

    // ---- weight pre-pass: fp32 -> f16, transposed -------------------------
    for (int L = 0; L < 6; ++L) {
        launch_transpose(P(L, 0),  WTl[L][0], D,   D,   stream);
        launch_transpose(P(L, 2),  WTl[L][1], D,   D,   stream);
        launch_transpose(P(L, 4),  WTl[L][2], D,   D,   stream);
        launch_transpose(P(L, 6),  WTl[L][3], D,   D,   stream);
        launch_transpose(P(L, 8),  WTl[L][4], D,   FFN, stream);
        launch_transpose(P(L, 10), WTl[L][5], FFN, D,   stream);
    }
    launch_transpose(Wout, WoutT, D, NVOCAB, stream);

    // ---- embedding + positional encoding ----------------------------------
    embed_kernel<<<(ROWS * D) / 256, 256, 0, stream>>>(tokens, emb, x, xh);

    // ---- transformer blocks (post-norm) -----------------------------------
    for (int L = 0; L < 6; ++L) {
        const float *bq = P(L, 1), *bk = P(L, 3), *bv = P(L, 5), *bo = P(L, 7);
        const float *b1 = P(L, 9), *b2 = P(L, 11);
        const float *g1 = P(L, 12), *be1 = P(L, 13);
        const float *g2 = P(L, 14), *be2 = P(L, 15);

        run_gemm<D, D, false, false>(xh, WTl[L][0], bq, xq, nullptr, stream);
        run_gemm<D, D, false, false>(xh, WTl[L][1], bk, xk, nullptr, stream);
        run_gemm<D, D, false, false>(xh, WTl[L][2], bv, xv, nullptr, stream);

        attention_kernel<<<8 * NHEAD, TSEQ, 0, stream>>>(xq, xk, xv, aoh);

        run_gemm<D, D, false, false>(aoh, WTl[L][3], bo, tmp, nullptr, stream);
        add_ln_kernel<<<ROWS / 8, 256, 0, stream>>>(x, tmp, g1, be1, x, xh);

        run_gemm<D, FFN, true, true>(xh, WTl[L][4], b1, nullptr, h1h, stream);
        run_gemm<FFN, D, false, false>(h1h, WTl[L][5], b2, tmp, nullptr, stream);
        add_ln_kernel<<<ROWS / 8, 256, 0, stream>>>(x, tmp, g2, be2, x, xh);
    }

    // ---- final LayerNorm + logits projection ------------------------------
    add_ln_kernel<<<ROWS / 8, 256, 0, stream>>>(x, nullptr, lnf_g, lnf_b, tmp, xfh);
    run_gemm<D, NVOCAB, false, false>(xfh, WoutT, bout, logits, nullptr, stream);
}
